// Memory_5952824673094
// MI455X (gfx1250) — matrix operand order repaired and checked by reading, not compiled, not run on hardware
//
#include <hip/hip_runtime.h>
#include <hip/hip_bf16.h>
#include <stdint.h>

// C[8192,8192] = A[8192,256] * Mem[8192,256]^T  via bf16x3 split-precision WMMA
// (hi*hi + hi*lo + lo*hi, f32 accumulate; dropped lo*lo term is ~2^-32 rel)
//
// Fast path: pre-split A/Mem into four bf16 planes in d_ws, then a WMMA GEMM
// with double-buffered LDS fed by GLOBAL_LOAD_ASYNC_TO_LDS_B128 (no VGPR
// transit, ASYNCcnt-tracked, overlapped with the WMMA block).
// Fallback path (ws too small): round-1 style in-loop split.

typedef __attribute__((ext_vector_type(16))) __bf16 v16bf;
typedef __attribute__((ext_vector_type(8)))  float  v8f;
typedef __attribute__((ext_vector_type(4)))  int    v4i;
typedef unsigned short u16;

#define F_DIM 256
#define N_DIM 8192
#define KT    32
#define LDP   40            // padded LDS row stride: 80B rows, 16B-aligned b128, conflict-free banks
#define PL    (128 * LDP)   // elements per plane tile (5120)
#define BUFE  (4 * PL)      // elements per buffer set (4 planes)

union Frag { uint4 q[2]; v16bf v; };

#if __has_builtin(__builtin_amdgcn_global_load_async_to_lds_b128) && \
    __has_builtin(__builtin_amdgcn_s_wait_asynccnt)
#define ASYNC_COPY 1
#endif

__device__ __forceinline__ void cp16(u16* lds, const u16* g) {
#ifdef ASYNC_COPY
    __builtin_amdgcn_global_load_async_to_lds_b128(
        (__attribute__((address_space(1))) v4i*)g,
        (__attribute__((address_space(3))) v4i*)lds,
        0, 0);
#else
    *(uint4*)lds = *(const uint4*)g;   // sync fallback: short-lived regs, no spill
#endif
}

__device__ __forceinline__ void wait_async0() {
#ifdef ASYNC_COPY
    __builtin_amdgcn_s_wait_asynccnt(0);
#endif
}

// ---------------------------------------------------------------------------
// Pre-pass: split f32 -> bf16 hi/lo planes (truncation split).
// ---------------------------------------------------------------------------
__global__ __launch_bounds__(256)
void split_bf16x2(const float* __restrict__ src,
                  u16* __restrict__ hiOut, u16* __restrict__ loOut)
{
    const int i = blockIdx.x * 256 + threadIdx.x;       // float4 index
    const float4 v = ((const float4*)src)[i];
    const float f[4] = {v.x, v.y, v.z, v.w};
    uint32_t h[4], l[4];
#pragma unroll
    for (int e = 0; e < 4; ++e) {
        uint32_t u  = __float_as_uint(f[e]);
        float    hi = __uint_as_float(u & 0xFFFF0000u);
        uint32_t ul = __float_as_uint(f[e] - hi);
        h[e] = u >> 16;
        l[e] = ul >> 16;
    }
    uint2 p;
    p.x = h[0] | (h[1] << 16); p.y = h[2] | (h[3] << 16);
    ((uint2*)hiOut)[i] = p;
    p.x = l[0] | (l[1] << 16); p.y = l[2] | (l[3] << 16);
    ((uint2*)loOut)[i] = p;
}

// ---------------------------------------------------------------------------
// Fast-path GEMM on pre-split bf16 planes, async double-buffered LDS.
// Block tile 128x128, 8 wave32s in 4(M) x 2(N); wave tile 32x64 = 2x4 WMMAs.
// ---------------------------------------------------------------------------
__global__ __launch_bounds__(256, 1)
void gemm_planes_async(const u16* __restrict__ aHiG, const u16* __restrict__ aLoG,
                       const u16* __restrict__ bHiG, const u16* __restrict__ bLoG,
                       float* __restrict__ Out)
{
    __shared__ u16 sm[2 * BUFE];    // 2 x (4 planes x 128 x LDP) = 80 KB

    const int tid   = threadIdx.x;
    const int lane  = tid & 31;
    const int wave  = tid >> 5;
    const int waveM = wave & 3;
    const int waveN = wave >> 2;

    const int blockM = blockIdx.y * 128;
    const int blockN = blockIdx.x * 128;

    v8f acc[2][4];
#pragma unroll
    for (int i = 0; i < 2; ++i)
#pragma unroll
        for (int j = 0; j < 4; ++j)
            acc[i][j] = (v8f){0.f,0.f,0.f,0.f,0.f,0.f,0.f,0.f};

    const int r16   = lane & 15;
    const int half  = lane >> 4;
    const int aKoff = half * 8;     // A frag: lanes 0-15 K{0..7,16..23}, 16-31 K{8..15,24..31}
    const int bKoff = half * 8;     // B frag: same K order as A frag

    // Staging map: per plane, tile = 128 rows x 32 bf16 = 512 x 16B; 2 per thread.
    int gOffA[2], gOffB[2], lOff[2];
#pragma unroll
    for (int i = 0; i < 2; ++i) {
        const int idx = tid + i * 256;
        const int row = idx >> 2;           // 4 x 16B per 32-elem row
        const int c8  = (idx & 3) * 8;
        gOffA[i] = (blockM + row) * F_DIM + c8;
        gOffB[i] = (blockN + row) * F_DIM + c8;
        lOff[i]  = row * LDP + c8;          // 16B-aligned
    }

    auto stage = [&](int bufOff, int k0) {
#pragma unroll
        for (int i = 0; i < 2; ++i) {
            cp16(&sm[bufOff + 0 * PL + lOff[i]], aHiG + gOffA[i] + k0);
            cp16(&sm[bufOff + 1 * PL + lOff[i]], aLoG + gOffA[i] + k0);
            cp16(&sm[bufOff + 2 * PL + lOff[i]], bHiG + gOffB[i] + k0);
            cp16(&sm[bufOff + 3 * PL + lOff[i]], bLoG + gOffB[i] + k0);
        }
    };

    stage(0, 0);
    wait_async0();
    __syncthreads();

#pragma unroll
    for (int kk = 0; kk < F_DIM / KT; ++kk) {
        const u16* cur = sm + (kk & 1) * BUFE;

        if (kk + 1 < F_DIM / KT)            // prefetch next tile into other buffer
            stage(((kk + 1) & 1) * BUFE, (kk + 1) * KT);

        // A fragments (hi/lo) for this wave's two 16-row subtiles
        Frag aHiF[2], aLoF[2];
#pragma unroll
        for (int mi = 0; mi < 2; ++mi) {
            const int base = (waveM * 32 + mi * 16 + r16) * LDP + aKoff;
            aHiF[mi].q[0] = *(const uint4*)&cur[0 * PL + base];
            aHiF[mi].q[1] = *(const uint4*)&cur[0 * PL + base + 16];
            aLoF[mi].q[0] = *(const uint4*)&cur[1 * PL + base];
            aLoF[mi].q[1] = *(const uint4*)&cur[1 * PL + base + 16];
        }

#pragma unroll
        for (int ni = 0; ni < 4; ++ni) {
            const int base = (waveN * 64 + ni * 16 + r16) * LDP + bKoff;
            Frag bHiF, bLoF;
            bHiF.q[0] = *(const uint4*)&cur[2 * PL + base];
            bHiF.q[1] = *(const uint4*)&cur[2 * PL + base + 16];
            bLoF.q[0] = *(const uint4*)&cur[3 * PL + base];
            bLoF.q[1] = *(const uint4*)&cur[3 * PL + base + 16];
#pragma unroll
            for (int mi = 0; mi < 2; ++mi) {
                acc[mi][ni] = __builtin_amdgcn_wmma_f32_16x16x32_bf16(
                    false, aHiF[mi].v, false, bHiF.v, (short)0, acc[mi][ni], false, false);
                acc[mi][ni] = __builtin_amdgcn_wmma_f32_16x16x32_bf16(
                    false, aHiF[mi].v, false, bLoF.v, (short)0, acc[mi][ni], false, false);
                acc[mi][ni] = __builtin_amdgcn_wmma_f32_16x16x32_bf16(
                    false, aLoF[mi].v, false, bHiF.v, (short)0, acc[mi][ni], false, false);
            }
        }

        wait_async0();                      // next tile landed (issued before compute)
        __syncthreads();                    // visible to all waves; orders buffer reuse
    }

    // Epilogue: D layout — VGPR v -> M = v + 8*(lane>=16), N = lane&15
    const int colBase = blockN + waveN * 64 + r16;
#pragma unroll
    for (int mi = 0; mi < 2; ++mi) {
        const int rowBase = blockM + waveM * 32 + mi * 16 + half * 8;
#pragma unroll
        for (int ni = 0; ni < 4; ++ni) {
            const int col = colBase + ni * 16;
#pragma unroll
            for (int v = 0; v < 8; ++v) {
                Out[(size_t)(rowBase + v) * N_DIM + col] = acc[mi][ni][v];
            }
        }
    }
}

// ---------------------------------------------------------------------------
// Fallback: all-in-one kernel (in-loop split), used if ws too small.
// ---------------------------------------------------------------------------
__global__ __launch_bounds__(256)
void gemm_bt_bf16x3(const float* __restrict__ A,
                    const float* __restrict__ Mem,
                    float* __restrict__ Out)
{
    __shared__ u16 aHi[PL];
    __shared__ u16 aLo[PL];
    __shared__ u16 bHi[PL];
    __shared__ u16 bLo[PL];

    const int tid   = threadIdx.x;
    const int lane  = tid & 31;
    const int wave  = tid >> 5;
    const int waveM = wave & 3;
    const int waveN = wave >> 2;

    const int blockM = blockIdx.y * 128;
    const int blockN = blockIdx.x * 128;

    v8f acc[2][4];
#pragma unroll
    for (int i = 0; i < 2; ++i)
#pragma unroll
        for (int j = 0; j < 4; ++j)
            acc[i][j] = (v8f){0.f,0.f,0.f,0.f,0.f,0.f,0.f,0.f};

    const int r16   = lane & 15;
    const int half  = lane >> 4;
    const int aKoff = half * 8;
    const int bKoff = half * 8;

    for (int k0 = 0; k0 < F_DIM; k0 += KT) {
        __syncthreads();
#pragma unroll
        for (int i = 0; i < 4; ++i) {
            const int idx = tid + i * 256;
            const int row = idx >> 3;
            const int c4  = (idx & 7) * 4;
            const float4 av = *(const float4*)(A   + (size_t)(blockM + row) * F_DIM + k0 + c4);
            const float4 bv = *(const float4*)(Mem + (size_t)(blockN + row) * F_DIM + k0 + c4);
            const float af[4] = {av.x, av.y, av.z, av.w};
            const float bf[4] = {bv.x, bv.y, bv.z, bv.w};
            uint32_t ha[4], la[4], hb[4], lb[4];
#pragma unroll
            for (int e = 0; e < 4; ++e) {
                uint32_t u  = __float_as_uint(af[e]);
                float    hi = __uint_as_float(u & 0xFFFF0000u);
                uint32_t ul = __float_as_uint(af[e] - hi);
                ha[e] = u >> 16;  la[e] = ul >> 16;
                u  = __float_as_uint(bf[e]);
                hi = __uint_as_float(u & 0xFFFF0000u);
                ul = __float_as_uint(bf[e] - hi);
                hb[e] = u >> 16;  lb[e] = ul >> 16;
            }
            const int ls = row * LDP + c4;
            uint2 p;
            p.x = ha[0] | (ha[1] << 16); p.y = ha[2] | (ha[3] << 16);
            *(uint2*)&aHi[ls] = p;
            p.x = la[0] | (la[1] << 16); p.y = la[2] | (la[3] << 16);
            *(uint2*)&aLo[ls] = p;
            p.x = hb[0] | (hb[1] << 16); p.y = hb[2] | (hb[3] << 16);
            *(uint2*)&bHi[ls] = p;
            p.x = lb[0] | (lb[1] << 16); p.y = lb[2] | (lb[3] << 16);
            *(uint2*)&bLo[ls] = p;
        }
        __syncthreads();

        Frag aHiF[2], aLoF[2];
#pragma unroll
        for (int mi = 0; mi < 2; ++mi) {
            const int base = (waveM * 32 + mi * 16 + r16) * LDP + aKoff;
            aHiF[mi].q[0] = *(const uint4*)&aHi[base];
            aHiF[mi].q[1] = *(const uint4*)&aHi[base + 16];
            aLoF[mi].q[0] = *(const uint4*)&aLo[base];
            aLoF[mi].q[1] = *(const uint4*)&aLo[base + 16];
        }
#pragma unroll
        for (int ni = 0; ni < 4; ++ni) {
            const int base = (waveN * 64 + ni * 16 + r16) * LDP + bKoff;
            Frag bHiF, bLoF;
            bHiF.q[0] = *(const uint4*)&bHi[base];
            bHiF.q[1] = *(const uint4*)&bHi[base + 16];
            bLoF.q[0] = *(const uint4*)&bLo[base];
            bLoF.q[1] = *(const uint4*)&bLo[base + 16];
#pragma unroll
            for (int mi = 0; mi < 2; ++mi) {
                acc[mi][ni] = __builtin_amdgcn_wmma_f32_16x16x32_bf16(
                    false, aHiF[mi].v, false, bHiF.v, (short)0, acc[mi][ni], false, false);
                acc[mi][ni] = __builtin_amdgcn_wmma_f32_16x16x32_bf16(
                    false, aHiF[mi].v, false, bLoF.v, (short)0, acc[mi][ni], false, false);
                acc[mi][ni] = __builtin_amdgcn_wmma_f32_16x16x32_bf16(
                    false, aLoF[mi].v, false, bHiF.v, (short)0, acc[mi][ni], false, false);
            }
        }
    }

    const int colBase = blockN + waveN * 64 + r16;
#pragma unroll
    for (int mi = 0; mi < 2; ++mi) {
        const int rowBase = blockM + waveM * 32 + mi * 16 + half * 8;
#pragma unroll
        for (int ni = 0; ni < 4; ++ni) {
            const int col = colBase + ni * 16;
#pragma unroll
            for (int v = 0; v < 8; ++v) {
                Out[(size_t)(rowBase + v) * N_DIM + col] = acc[mi][ni][v];
            }
        }
    }
}

extern "C" void kernel_launch(void* const* d_in, const int* in_sizes, int n_in,
                              void* d_out, int out_size, void* d_ws, size_t ws_size,
                              hipStream_t stream) {
    // setup_inputs order: inputs[8192,256] f32, targets (unused), mem[8192,256] f32, epoch (unused)
    const float* A   = (const float*)d_in[0];
    const float* Mem = (const float*)d_in[2];
    float*       Out = (float*)d_out;

    const size_t planeElems = (size_t)N_DIM * F_DIM;    // 2M elements
    const size_t planeBytes = planeElems * sizeof(u16); // 4 MB

    dim3 grid(N_DIM / 128, N_DIM / 128);                // 64 x 64
    dim3 block(256);

    if (ws_size >= 4 * planeBytes) {
        u16* aHi = (u16*)d_ws;
        u16* aLo = aHi + planeElems;
        u16* bHi = aLo + planeElems;
        u16* bLo = bHi + planeElems;

        const int nvec = (int)(planeElems / 4);         // float4 count per matrix
        split_bf16x2<<<nvec / 256, 256, 0, stream>>>(A,   aHi, aLo);
        split_bf16x2<<<nvec / 256, 256, 0, stream>>>(Mem, bHi, bLo);
        gemm_planes_async<<<grid, block, 0, stream>>>(aHi, aLo, bHi, bLo, Out);
    } else {
        gemm_bt_bf16x3<<<grid, block, 0, stream>>>(A, Mem, Out);
    }
}
